// SE3Decoder_74062416053479
// MI455X (gfx1250) — matrix operand order repaired and checked by reading, not compiled, not run on hardware
//
#include <hip/hip_runtime.h>
#include <hip/hip_bf16.h>
#include <math.h>

// Problem constants (match reference)
#define N_IONS 64
#define N_GRIDP 4096
#define DEG 8
#define E_TOT (N_GRIDP * DEG)     // 32768
#define HIDDEN 16
#define LDOS 64
#define C_KV 128                  // 2*LDOS
#define MID 32
#define KPRIME 128                // combined contraction length: 32 (r00) + 3*32 (r10)

typedef __attribute__((ext_vector_type(16))) _Float16 v16h;
typedef __attribute__((ext_vector_type(8)))  float    v8f;
typedef __attribute__((ext_vector_type(8)))  _Float16 h8v;   // 16-byte f16 packet

// ---------------------------------------------------------------------------
// K1: per-ion reduced weights  Wc[s][o][k']  (f16, 64*128*128 = 2 MB)
//   k' = c*32 + k
//   c==0 : sum_i w3_00[(o*16+i)*32+k] * node0[s,i]
//   c>=1 : m=c-1 : sum_i w3_10[(o*16+i)*32+k] * node1[s,i,m]
// ---------------------------------------------------------------------------
__global__ void k_prep(const float* __restrict__ node0,
                       const float* __restrict__ node1,
                       const float* __restrict__ w3_00,
                       const float* __restrict__ w3_10,
                       _Float16* __restrict__ Wc) {
  const int s = blockIdx.x;
  __shared__ float n0s[16];
  __shared__ float n1s[48];
  const int t = threadIdx.x;
  if (t < 16) n0s[t] = node0[s * 16 + t];
  if (t < 48) n1s[t] = node1[s * 48 + t];
  __syncthreads();
  for (int idx = t; idx < C_KV * KPRIME; idx += blockDim.x) {
    const int o  = idx >> 7;
    const int kp = idx & 127;
    const int k  = kp & 31;
    const int c  = kp >> 5;
    float acc = 0.f;
    if (c == 0) {
#pragma unroll
      for (int i = 0; i < 16; ++i)
        acc = fmaf(w3_00[(o * 16 + i) * 32 + k], n0s[i], acc);
    } else {
      const int m = c - 1;
#pragma unroll
      for (int i = 0; i < 16; ++i)
        acc = fmaf(w3_10[(o * 16 + i) * 32 + k], n1s[i * 3 + m], acc);
    }
    Wc[((size_t)s * C_KV + o) * KPRIME + kp] = (_Float16)acc;
  }
}

// ---------------------------------------------------------------------------
// K2: per-16-edge tile.  Layer1 (VALU) -> layer2 via v_wmma_f32_16x16x32_f16
// -> basis-folded hidden hh[e][0..127] -> kv[e][o] = hh[e] . Wc[src[e]][o]
// Wave w owns hh column block w (branch-free basis folding).
// ---------------------------------------------------------------------------
__global__ void k_edges(const float* __restrict__ edge_feat,
                        const float* __restrict__ basis00,
                        const float* __restrict__ basis10,
                        const int*   __restrict__ src,
                        const float* __restrict__ w1a, const float* __restrict__ b1a,
                        const float* __restrict__ w2a, const float* __restrict__ b2a,
                        const float* __restrict__ w1b, const float* __restrict__ b1b,
                        const float* __restrict__ w2b, const float* __restrict__ b2b,
                        const _Float16* __restrict__ Wc,
                        float* __restrict__ kv) {
  __shared__ __align__(16) _Float16 hA[2][16][32];  // layer1 acts (WMMA A), f16
  __shared__ __align__(16) float    hh[16][KPRIME]; // basis-folded hidden, f32
  __shared__ float    invs[16];
  __shared__ float    scl[4][16];      // [col-block][edge] basis scale
  __shared__ int      srcs[16];

  const int e0 = blockIdx.x * 16;
  const int t  = threadIdx.x;

  if (t < 16) {
    invs[t]    = edge_feat[e0 + t];
    scl[0][t]  = basis00[e0 + t];
    scl[1][t]  = basis10[(e0 + t) * 3 + 0];
    scl[2][t]  = basis10[(e0 + t) * 3 + 1];
    scl[3][t]  = basis10[(e0 + t) * 3 + 2];
    srcs[t]    = src[e0 + t];
  }
  __syncthreads();

  // layer 1 (1 -> 32), both MLPs: h1[e][k] = relu(inv*w1[k] + b1[k])
  for (int idx = t; idx < 16 * 32; idx += 128) {
    const int e = idx >> 5, k = idx & 31;
    const float x = invs[e];
    hA[0][e][k] = (_Float16)fmaxf(fmaf(x, w1a[k], b1a[k]), 0.f);
    hA[1][e][k] = (_Float16)fmaxf(fmaf(x, w1b[k], b1b[k]), 0.f);
  }
  __syncthreads();

  // layer 2 (32 -> 32) via WMMA. Wave w computes hh columns [w*32, w*32+32):
  //   w==0  -> b00  * h00
  //   w>=1  -> b10[w-1] * h10
  {
    const int wv  = t >> 5;
    const int l   = t & 31;
    const int mlp = (wv > 0) ? 1 : 0;
    const float* w2 = mlp ? w2b : w2a;
    const float* b2 = mlp ? b2b : b2a;

    // A fragment (shared by both N halves)
    v16h a;
    {
      const int m  = l & 15;
      const int kb = (l >> 4) * 8;        // lanes 16-31 hold K+8
#pragma unroll
      for (int idx = 0; idx < 16; ++idx) {
        const int K = idx + ((idx >= 8) ? 8 : 0) + kb;   // idx>=8 -> K+16 block
        a[idx] = hA[mlp][m][K];
      }
    }
#pragma unroll
    for (int nhalf = 0; nhalf < 32; nhalf += 16) {
      v16h b;
      {
        const int n  = (l & 15) + nhalf;  // B[K][N] = w2[n*32 + K]
        const int kb = (l >= 16) ? 8 : 0;
#pragma unroll
        for (int idx = 0; idx < 16; ++idx)
          b[idx] = (_Float16)w2[n * 32 + (idx + ((idx >= 8) ? 8 : 0) + kb)];
      }
      v8f c = {};
      c = __builtin_amdgcn_wmma_f32_16x16x32_f16(false, a, false, b,
                                                 (short)0, c, false, false);
      const int n    = (l & 15) + nhalf;
      const float bi = b2[n];
#pragma unroll
      for (int vi = 0; vi < 8; ++vi) {
        const int m = vi + ((l >= 16) ? 8 : 0);           // edge row
        const float hv = fmaxf(c[vi] + bi, 0.f);
        hh[m][wv * 32 + n] = scl[wv][m] * hv;             // branch-free fold
      }
    }
  }
  __syncthreads();

  // kv[e][o] = sum_{k'} hh[e][k'] * Wc[src[e]][o][k']  (o = thread id)
  // 16B vector loads: h8v from global (f16), float4 broadcast from LDS.
  const int o = t;
  for (int e = 0; e < 16; ++e) {
    const h8v* wr =
        (const h8v*)(Wc + ((size_t)(srcs[e]) * C_KV + o) * KPRIME);
    __builtin_prefetch(wr, 0, 1);
    const float4* hr4 = (const float4*)(&hh[e][0]);
    float acc = 0.f;
#pragma unroll
    for (int kk = 0; kk < KPRIME / 8; ++kk) {
      const h8v  w8 = wr[kk];
      const float4 ha = hr4[2 * kk + 0];
      const float4 hb = hr4[2 * kk + 1];
      acc = fmaf((float)w8[0], ha.x, acc);
      acc = fmaf((float)w8[1], ha.y, acc);
      acc = fmaf((float)w8[2], ha.z, acc);
      acc = fmaf((float)w8[3], ha.w, acc);
      acc = fmaf((float)w8[4], hb.x, acc);
      acc = fmaf((float)w8[5], hb.y, acc);
      acc = fmaf((float)w8[6], hb.z, acc);
      acc = fmaf((float)w8[7], hb.w, acc);
    }
    kv[(size_t)(e0 + e) * C_KV + o] = acc;
  }
}

// ---------------------------------------------------------------------------
// K3: per-16-grid-point tile.  q, logits, 8-way softmax, z, then the
// (16x80pad96) @ (96x64) projection via WMMA.
// ---------------------------------------------------------------------------
__global__ void k_grid(const float* __restrict__ node0,
                       const float* __restrict__ kv,
                       const float* __restrict__ w_q,
                       const float* __restrict__ w_proj,
                       float* __restrict__ out) {
  __shared__ __align__(16) float    qs[16][LDOS];  // 4 KB
  __shared__ float    lo[128];
  __shared__ float    wexp[128];
  __shared__ __align__(16) _Float16 zc[16][96];    // 3 KB, K padded 80 -> 96

  const int g0 = blockIdx.x * 16;
  const int t  = threadIdx.x;

  // q[g][o] = sum_i w_q[o][i] * node0[64+g][i]
  for (int idx = t; idx < 16 * LDOS; idx += 128) {
    const int lg = idx >> 6, o = idx & 63;
    const float4* nd =
        (const float4*)(node0 + (size_t)(N_IONS + g0 + lg) * 16);
    const float4* wq = (const float4*)(w_q + o * 16);
    float acc = 0.f;
#pragma unroll
    for (int i4 = 0; i4 < 4; ++i4) {
      const float4 n4 = nd[i4], w4 = wq[i4];
      acc = fmaf(w4.x, n4.x, acc);
      acc = fmaf(w4.y, n4.y, acc);
      acc = fmaf(w4.z, n4.z, acc);
      acc = fmaf(w4.w, n4.w, acc);
    }
    qs[lg][o] = acc;
  }
  __syncthreads();

  // logits: one per edge (128 edges per block); key = kv[:, 64:]
  {
    const int lg = t >> 3;
    const float4* key =
        (const float4*)(kv + ((size_t)g0 * DEG + t) * C_KV + LDOS);
    const float4* qv = (const float4*)(&qs[lg][0]);
    float acc = 0.f;
#pragma unroll
    for (int o4 = 0; o4 < LDOS / 4; ++o4) {
      const float4 k4 = key[o4], q4 = qv[o4];
      acc = fmaf(k4.x, q4.x, acc);
      acc = fmaf(k4.y, q4.y, acc);
      acc = fmaf(k4.z, q4.z, acc);
      acc = fmaf(k4.w, q4.w, acc);
    }
    lo[t] = acc * 0.125f;                         // 1/sqrt(LDOS)
  }
  __syncthreads();

  // softmax weights over each group of 8 edges (redundant per-thread scan)
  {
    const int base = (t >> 3) << 3;
    float m = lo[base];
#pragma unroll
    for (int j = 1; j < 8; ++j) m = fmaxf(m, lo[base + j]);
    float den = 0.f;
#pragma unroll
    for (int j = 0; j < 8; ++j) den += __expf(lo[base + j] - m);
    wexp[t] = __expf(lo[t] - m) / den;
  }
  __syncthreads();

  // z[g][o] = sum_j wexp * value[e][o];  zc = [z | node0[64+g] | 0-pad]
  for (int idx = t; idx < 16 * LDOS; idx += 128) {
    const int lg = idx >> 6, o = idx & 63;
    const float* val = kv + ((size_t)(g0 + lg) * DEG) * C_KV + o;
    float acc = 0.f;
#pragma unroll
    for (int j = 0; j < 8; ++j) acc = fmaf(wexp[lg * 8 + j], val[j * C_KV], acc);
    zc[lg][o] = (_Float16)acc;
  }
  for (int idx = t; idx < 16 * 32; idx += 128) {
    const int lg = idx >> 5, i = idx & 31;
    zc[lg][64 + i] =
        (i < 16) ? (_Float16)node0[(size_t)(N_IONS + g0 + lg) * 16 + i]
                 : (_Float16)0.f;
  }
  __syncthreads();

  // projection: out[g][n] = sum_{j<80} zc[g][j] * w_proj[n*80+j]
  const int wv = t >> 5;
  const int l  = t & 31;
  const int nb = wv * 16;                // each wave owns one 16-wide N tile
  v8f acc = {};
#pragma unroll
  for (int kc = 0; kc < 3; ++kc) {
    v16h a, b;
    {
      const int m   = l & 15;
      const int kb8 = (l >> 4) * 8;
#pragma unroll
      for (int idx = 0; idx < 16; ++idx) {
        const int K = kc * 32 + idx + ((idx >= 8) ? 8 : 0) + kb8;
        a[idx] = zc[m][K];
      }
    }
    {
      const int n    = (l & 15) + nb;
      const int kb16 = (l >= 16) ? 8 : 0;
#pragma unroll
      for (int idx = 0; idx < 16; ++idx) {
        const int j = kc * 32 + idx + ((idx >= 8) ? 8 : 0) + kb16;
        b[idx] = (j < 80) ? (_Float16)w_proj[n * 80 + j] : (_Float16)0.f;
      }
    }
    acc = __builtin_amdgcn_wmma_f32_16x16x32_f16(false, a, false, b,
                                                 (short)0, acc, false, false);
  }
  {
    const int n = (l & 15) + nb;
#pragma unroll
    for (int vi = 0; vi < 8; ++vi) {
      const int m = vi + ((l >= 16) ? 8 : 0);
      out[(size_t)(g0 + m) * LDOS + n] = acc[vi];
    }
  }
}

// ---------------------------------------------------------------------------
extern "C" void kernel_launch(void* const* d_in, const int* in_sizes, int n_in,
                              void* d_out, int out_size, void* d_ws, size_t ws_size,
                              hipStream_t stream) {
  const float* node0   = (const float*)d_in[0];
  const float* node1   = (const float*)d_in[1];
  const float* edge_f  = (const float*)d_in[2];
  const float* basis00 = (const float*)d_in[3];
  const float* basis10 = (const float*)d_in[4];
  const float* r00_w1  = (const float*)d_in[5];
  const float* r00_b1  = (const float*)d_in[6];
  const float* r00_w2  = (const float*)d_in[7];
  const float* r00_b2  = (const float*)d_in[8];
  const float* r00_w3  = (const float*)d_in[9];
  const float* r10_w1  = (const float*)d_in[10];
  const float* r10_b1  = (const float*)d_in[11];
  const float* r10_w2  = (const float*)d_in[12];
  const float* r10_b2  = (const float*)d_in[13];
  const float* r10_w3  = (const float*)d_in[14];
  const float* w_q     = (const float*)d_in[15];
  const float* w_proj  = (const float*)d_in[16];
  const int*   src     = (const int*)d_in[17];
  (void)in_sizes; (void)n_in; (void)out_size; (void)ws_size;

  char* ws = (char*)d_ws;
  _Float16* Wc = (_Float16*)ws;                                   // 2 MB
  float* kv = (float*)(ws + (size_t)N_IONS * C_KV * KPRIME * sizeof(_Float16)); // 16 MB

  k_prep<<<N_IONS, 256, 0, stream>>>(node0, node1, r00_w3, r10_w3, Wc);
  k_edges<<<E_TOT / 16, 128, 0, stream>>>(edge_f, basis00, basis10, src,
                                          r00_w1, r00_b1, r00_w2, r00_b2,
                                          r10_w1, r10_b1, r10_w2, r10_b2,
                                          Wc, kv);
  k_grid<<<N_GRIDP / 16, 128, 0, stream>>>(node0, kv, w_q, w_proj, (float*)d_out);
}
